// BasicBlock_85684597555725
// MI455X (gfx1250) — matrix operand order repaired and checked by reading, not compiled, not run on hardware
//
#include <hip/hip_runtime.h>
#include <hip/hip_bf16.h>
#include <stdint.h>

// ---------------------------------------------------------------------------
// Fused MoE-BasicBlock for MI455X (gfx1250, wave32, WMMA).
//   conv3x3(x, sum_e a[e]*w1[e]) -> BN1 -> ReLU -> conv3x3 -> BN2 -> +x -> ReLU
// Implicit GEMM with V_WMMA_F32_16X16X32_BF16 (f32 accumulate).
// Wave tiling: 2 c_out groups x 2 row-tiles per wave -> 4 independent WMMA
// chains, 1:1 ds_load_b128 : wmma ratio.
// ---------------------------------------------------------------------------

typedef __attribute__((ext_vector_type(16))) __bf16 v16bf;
typedef __attribute__((ext_vector_type(8)))  float  v8f;

struct U32x8 { uint4 lo, hi; };

#define Hc 112
#define Wc 112
#define PLANE (Hc * Wc)            // 12544
#define TILE_DW (10 * 18 * 36)     // halo tile: 10 rows x 18 cols x 32 cpairs (pitch 36)

#if defined(__has_builtin)
#if __has_builtin(__builtin_amdgcn_global_load_async_to_lds_b32)
#define USE_ASYNC_LDS 1
#endif
#endif

#ifdef USE_ASYNC_LDS
typedef __attribute__((address_space(1))) int* as1ptr;   // "__device__ int*"
typedef __attribute__((address_space(3))) int* as3ptr;   // "__shared__ int*"
#endif

static __device__ __forceinline__ unsigned int bf16pack(float a, float b) {
  // round-to-nearest-even f32 -> bf16, packed {b[hi], a[lo]}
  unsigned int ua = __float_as_uint(a), ub = __float_as_uint(b);
  ua += 0x7FFFu + ((ua >> 16) & 1u);
  ub += 0x7FFFu + ((ub >> 16) & 1u);
  return (ua >> 16) | (ub & 0xFFFF0000u);
}

static __device__ __forceinline__ v16bf as_bf(U32x8 u) {
  return __builtin_bit_cast(v16bf, u);
}

// ---------------------------------------------------------------------------
// Kernel 0: combine 10 experts with alpha and pre-swizzle into the exact
// WMMA A-fragment layout (ISA 7.12.2, 16-bit A 16x32).
// K ordering: k = (kh*3+kw)*64 + c_in  -> chunk kc: pos = kc>>1, cin base = (kc&1)*32
// frag layout: [(g*18 + kc)*32 + lane]*8 dwords (2x b128 per lane).
// ---------------------------------------------------------------------------
__global__ __launch_bounds__(32) void pack_weights(
    const float* __restrict__ w1, const float* __restrict__ w2,
    const float* __restrict__ alpha,
    unsigned int* __restrict__ frag1, unsigned int* __restrict__ frag2) {
  const int bi = blockIdx.x;   // 144 blocks: 72 per conv
  const int l  = threadIdx.x;  // 32 lanes
  const float* w = (bi < 72) ? w1 : w2;
  unsigned int* frag = (bi < 72) ? frag1 : frag2;
  const int lb = bi % 72;
  const int g  = lb / 18;      // c_out group 0..3
  const int kc = lb % 18;      // K chunk 0..17
  const int co = g * 16 + (l & 15);
  const int pos = kc >> 1;               // kh*3+kw
  const int cinb = (kc & 1) * 32;

  float av[10];
#pragma unroll
  for (int e = 0; e < 10; ++e) av[e] = alpha[e];

  unsigned int* dst = frag + ((size_t)((g * 18 + kc) * 32 + l)) * 8;
#pragma unroll
  for (int j = 0; j < 8; ++j) {
    const int Klo = (l < 16) ? ((j < 4) ? 2 * j : 16 + 2 * (j - 4))
                             : ((j < 4) ? 8 + 2 * j : 24 + 2 * (j - 4));
    const int ci0 = cinb + Klo;
    float s0 = 0.f, s1 = 0.f;
    for (int e = 0; e < 10; ++e) {
      const float* wb = w + ((size_t)(e * 64 + co) * 64) * 9 + pos;
      s0 += av[e] * wb[(size_t)ci0 * 9];
      s1 += av[e] * wb[(size_t)(ci0 + 1) * 9];
    }
    dst[j] = bf16pack(s0, s1);
  }
}

// ---------------------------------------------------------------------------
// Kernel 0b: fold BN params: scale = gamma*rsqrt(var+eps), bias = beta - mean*scale
// bnp layout: [s1(64) | b1(64) | s2(64) | b2(64)]
// ---------------------------------------------------------------------------
__global__ __launch_bounds__(64) void bn_prep(
    const float* __restrict__ g1, const float* __restrict__ be1,
    const float* __restrict__ m1, const float* __restrict__ v1,
    const float* __restrict__ g2, const float* __restrict__ be2,
    const float* __restrict__ m2, const float* __restrict__ v2,
    float* __restrict__ bnp) {
  const int c = threadIdx.x;
  const float s1 = g1[c] * rsqrtf(v1[c] + 1e-5f);
  const float s2 = g2[c] * rsqrtf(v2[c] + 1e-5f);
  bnp[c]       = s1;
  bnp[64 + c]  = be1[c] - m1[c] * s1;
  bnp[128 + c] = s2;
  bnp[192 + c] = be2[c] - m2[c] * s2;
}

// ---------------------------------------------------------------------------
// Conv core: block = (b, 8 rows, 16 cols) of output, all 64 channels.
// Wave wv: group-pair gp = wv&1 (c_out groups 2gp, 2gp+1), row-pair (wv>>1)*2.
// LDS tile: [row 0..9][col 0..17][cpair 0..31] dwords, pitch 36 (bank-tuned:
// the 16 lanes of a half-wave land on banks {0,4,...,60} for b128 loads).
// ---------------------------------------------------------------------------
#define CONV_DECODE()                                           \
  const int bi = blockIdx.x;                                    \
  const int wb = bi % 7, hb = (bi / 7) % 14, b = bi / 98;       \
  const int w0 = wb * 16, h0 = hb * 8;                          \
  const int tid = threadIdx.x;

#define CONV_CORE()                                                           \
  const int lane = tid & 31, wv = tid >> 5;                                   \
  const int gp = wv & 1;                                                      \
  const int t0 = (wv >> 1) * 2;                                               \
  const int g0 = gp * 2, g1 = g0 + 1;                                         \
  const int hi8 = (lane & 16) >> 1; /* 0 or 8 */                              \
  const int nl = lane & 15;                                                   \
  v8f acc00 = {}, acc01 = {}, acc10 = {}, acc11 = {};                         \
  const unsigned int* wf0 = frag + ((size_t)g0 * 18 * 32 + lane) * 8;         \
  const unsigned int* wf1 = frag + ((size_t)g1 * 18 * 32 + lane) * 8;         \
  for (int kc = 0; kc < 18; ++kc) {                                           \
    const int pos = kc >> 1;                                                  \
    const int kh = pos / 3, kw = pos - kh * 3;                                \
    const uint4* a0p = (const uint4*)(wf0 + kc * 256);                        \
    const uint4* a1p = (const uint4*)(wf1 + kc * 256);                        \
    U32x8 a0u; a0u.lo = a0p[0]; a0u.hi = a0p[1];                              \
    U32x8 a1u; a1u.lo = a1p[0]; a1u.hi = a1p[1];                              \
    const v16bf A0 = as_bf(a0u);                                              \
    const v16bf A1 = as_bf(a1u);                                              \
    const int cpb = ((kc & 1) << 4) + (hi8 >> 1);                             \
    const int base0 = ((t0 + kh) * 18 + (nl + kw)) * 36 + cpb;                \
    const uint4* p0 = (const uint4*)&lds[base0];                              \
    const uint4* p1 = (const uint4*)&lds[base0 + 18 * 36];                    \
    U32x8 bu0; bu0.lo = p0[0]; bu0.hi = p0[2];                                \
    U32x8 bu1; bu1.lo = p1[0]; bu1.hi = p1[2];                                \
    const v16bf B0 = as_bf(bu0);                                              \
    const v16bf B1 = as_bf(bu1);                                              \
    acc00 = __builtin_amdgcn_wmma_f32_16x16x32_bf16(                          \
        false, A0, false, B0, (short)0, acc00, false, false);                 \
    acc01 = __builtin_amdgcn_wmma_f32_16x16x32_bf16(                          \
        false, A0, false, B1, (short)0, acc01, false, false);                 \
    acc10 = __builtin_amdgcn_wmma_f32_16x16x32_bf16(                          \
        false, A1, false, B0, (short)0, acc10, false, false);                 \
    acc11 = __builtin_amdgcn_wmma_f32_16x16x32_bf16(                          \
        false, A1, false, B1, (short)0, acc11, false, false);                 \
  }

// conv1: x (f32 NCHW) -> BN1 -> ReLU -> bf16 pair-interleaved intermediate
__global__ __launch_bounds__(256) void conv1_kernel(
    const float* __restrict__ x, const unsigned int* __restrict__ frag,
    const float* __restrict__ bns, const float* __restrict__ bnb,
    unsigned int* __restrict__ inter) {
  __shared__ unsigned int lds[TILE_DW];
  CONV_DECODE();

  // stage halo tile: f32 x2 -> packed bf16 pair dword (conversion => VALU path)
  for (int i = tid; i < 32 * 10 * 18; i += 256) {
    const int col = i % 18;
    const int t2 = i / 18;
    const int row = t2 % 10;
    const int cp = t2 / 10;
    const int y = h0 - 1 + row, xx = w0 - 1 + col;
    unsigned int pk = 0u;
    if ((unsigned)y < Hc && (unsigned)xx < Wc) {
      const float* p = x + ((size_t)(b * 64 + 2 * cp) * Hc + y) * Wc + xx;
      pk = bf16pack(p[0], p[PLANE]);
    }
    lds[(row * 18 + col) * 36 + cp] = pk;
  }
  __syncthreads();

  CONV_CORE();

  // epilogue: BN1 + ReLU, pack channel pairs, store to intermediate
  const int wcol = w0 + nl;
  const v8f aa[4] = {acc00, acc01, acc10, acc11};
  const int gg[4] = {g0, g0, g1, g1};
  const int tt[4] = {0, 1, 0, 1};
#pragma unroll
  for (int q = 0; q < 4; ++q) {
    const int h = h0 + t0 + tt[q];
    const int cpb_out = gg[q] * 8 + (hi8 >> 1);
    const v8f a = aa[q];
#pragma unroll
    for (int r = 0; r < 8; r += 2) {
      const int cp = cpb_out + (r >> 1);
      const int c0 = 2 * cp;
      const float v0 = fmaxf(a[r]     * bns[c0]     + bnb[c0],     0.f);
      const float v1 = fmaxf(a[r + 1] * bns[c0 + 1] + bnb[c0 + 1], 0.f);
      inter[((size_t)(b * 32 + cp) * Hc + h) * Wc + wcol] = bf16pack(v0, v1);
    }
  }
}

// conv2: bf16 intermediate -> BN2 + residual(x) + ReLU -> f32 d_out (NCHW)
__global__ __launch_bounds__(256) void conv2_kernel(
    const unsigned int* __restrict__ inter, const unsigned int* __restrict__ frag,
    const float* __restrict__ bns, const float* __restrict__ bnb,
    const float* __restrict__ xres, float* __restrict__ out) {
  __shared__ unsigned int lds[TILE_DW];
  CONV_DECODE();

  // stage halo tile: pure dword copy -> use CDNA5 async global->LDS path
  for (int i = tid; i < 32 * 10 * 18; i += 256) {
    const int col = i % 18;
    const int t2 = i / 18;
    const int row = t2 % 10;
    const int cp = t2 / 10;
    const int y = h0 - 1 + row, xx = w0 - 1 + col;
    const int dst = (row * 18 + col) * 36 + cp;
#ifdef USE_ASYNC_LDS
    if ((unsigned)y < Hc && (unsigned)xx < Wc) {
      const unsigned int* src = inter + ((size_t)(b * 32 + cp) * Hc + y) * Wc + xx;
      __builtin_amdgcn_global_load_async_to_lds_b32(
          (as1ptr)const_cast<unsigned int*>(src), (as3ptr)&lds[dst], 0, 0);
    } else {
      lds[dst] = 0u;   // zero-pad border via normal ds store
    }
#else
    unsigned int pk = 0u;
    if ((unsigned)y < Hc && (unsigned)xx < Wc)
      pk = inter[((size_t)(b * 32 + cp) * Hc + y) * Wc + xx];
    lds[dst] = pk;
#endif
  }
#ifdef USE_ASYNC_LDS
#if __has_builtin(__builtin_amdgcn_s_wait_asynccnt)
  __builtin_amdgcn_s_wait_asynccnt(0);
#else
  asm volatile("s_wait_asynccnt 0x0" ::: "memory");
#endif
#endif
  __syncthreads();

  CONV_CORE();

  const int wcol = w0 + nl;
  const v8f aa[4] = {acc00, acc01, acc10, acc11};
  const int gg[4] = {g0, g0, g1, g1};
  const int tt[4] = {0, 1, 0, 1};
#pragma unroll
  for (int q = 0; q < 4; ++q) {
    const int h = h0 + t0 + tt[q];
    const v8f a = aa[q];
#pragma unroll
    for (int r = 0; r < 8; ++r) {
      const int c = gg[q] * 16 + hi8 + r;
      const size_t off = ((size_t)(b * 64 + c) * Hc + h) * Wc + wcol;
      const float v = a[r] * bns[c] + bnb[c] + xres[off];
      out[off] = fmaxf(v, 0.f);
    }
  }
}

// ---------------------------------------------------------------------------
// Host launch
// ---------------------------------------------------------------------------
extern "C" void kernel_launch(void* const* d_in, const int* in_sizes, int n_in,
                              void* d_out, int out_size, void* d_ws, size_t ws_size,
                              hipStream_t stream) {
  (void)in_sizes; (void)n_in; (void)out_size; (void)ws_size;
  const float* x     = (const float*)d_in[0];
  const float* w1    = (const float*)d_in[1];
  const float* w2    = (const float*)d_in[2];
  const float* alpha = (const float*)d_in[3];
  const float* g1  = (const float*)d_in[4];
  const float* be1 = (const float*)d_in[5];
  const float* m1  = (const float*)d_in[6];
  const float* v1  = (const float*)d_in[7];
  const float* g2  = (const float*)d_in[8];
  const float* be2 = (const float*)d_in[9];
  const float* m2  = (const float*)d_in[10];
  const float* v2  = (const float*)d_in[11];

  char* ws = (char*)d_ws;
  unsigned int* frag1 = (unsigned int*)(ws);            // 73,728 B
  unsigned int* frag2 = (unsigned int*)(ws + 73728);    // 73,728 B
  float*        bnp   = (float*)(ws + 147456);          // 1,024 B
  unsigned int* inter = (unsigned int*)(ws + 148480);   // 51,380,224 B (bf16 pairs)

  pack_weights<<<dim3(144), dim3(32), 0, stream>>>(w1, w2, alpha, frag1, frag2);
  bn_prep<<<dim3(1), dim3(64), 0, stream>>>(g1, be1, m1, v1, g2, be2, m2, v2, bnp);

  // 32 batches * 14 row-tiles * 7 col-tiles = 3136 blocks, 8 waves each
  conv1_kernel<<<dim3(3136), dim3(256), 0, stream>>>(x, frag1, bnp, bnp + 64, inter);
  conv2_kernel<<<dim3(3136), dim3(256), 0, stream>>>(inter, frag2, bnp + 128, bnp + 192,
                                                     x, (float*)d_out);
}
